// MultiScaleDeformableAttention_65051574665311
// MI455X (gfx1250) — matrix operand order repaired and checked by reading, not compiled, not run on hardware
//
#include <hip/hip_runtime.h>
#include <hip/hip_bf16.h>

#define B_        8
#define Q_        900
#define EMBED_    256
#define HEADS_    8
#define LEVELS_   4
#define POINTS_   4
#define HEAD_DIM_ 32
#define NUM_KEYS_ 13294

typedef __attribute__((ext_vector_type(16))) __bf16 v16bf;
typedef __attribute__((ext_vector_type(8)))  float  v8f;

// ---------------------------------------------------------------------------
// Pack fp32 W[K,N] into bf16 WMMA B-fragment order:
//   P[((kt*(N/16) + nt)*32 + lane)*16 + e] = bf16( W[kt*32 + (lane>>4)*16 + e][nt*16 + (lane&15)] )
// One wave per (kt, nt) fragment. Done once; consumed ~6647x by the value GEMM.
// ---------------------------------------------------------------------------
__global__ __launch_bounds__(32)
void pack_w_kernel(const float* __restrict__ W, __bf16* __restrict__ P, int N)
{
    const int lane = threadIdx.x & 31;
    const int hi   = lane >> 4;
    const int mr   = lane & 15;
    const int kt   = blockIdx.x;          // K/32 tiles
    const int nt   = blockIdx.y;          // N/16 tiles

    const float* wp = W + (size_t)(kt * 32 + hi * 16) * N + nt * 16 + mr;
    v16bf v;
    #pragma unroll
    for (int e = 0; e < 16; ++e) v[e] = (__bf16)wp[(size_t)e * N];

    const int ntiles = N >> 4;
    *(v16bf*)(P + (((size_t)kt * ntiles + nt) * 32 + lane) * 16) = v;
}

// ---------------------------------------------------------------------------
// C[M,N] = A[M,K] * W[K,N] + bias[N]; A fp32, W pre-packed bf16 fragments.
// One wave per 16(M) x 64(N) tile. blockIdx.x = N-tile (so blocks sharing the
// same A rows are adjacent in launch order -> A fetched from HBM once).
// out_mode 0: row-major C[m*N+n]
// out_mode 1: scatter to (B, HEADS, NUM_KEYS, 32) layout (value projection)
// ---------------------------------------------------------------------------
__global__ __launch_bounds__(32)
void gemm_wmma_bf16(const float* __restrict__ A, const __bf16* __restrict__ Wp,
                    const float* __restrict__ bias, float* __restrict__ C,
                    int N, int K, int out_mode)
{
    const int lane = threadIdx.x & 31;
    const int hi   = lane >> 4;
    const int mr   = lane & 15;
    const int n0   = blockIdx.x * 64;
    const int m0   = blockIdx.y * 16;
    const int ntiles = N >> 4;

    v8f acc[4] = {};

    for (int k0 = 0, kt = 0; k0 < K; k0 += 32, ++kt) {
        // A fragment: 16x32 bf16. Lane half 'hi' holds K = k0+16*hi..+15
        const float* ap = A + (size_t)(m0 + mr) * K + k0 + hi * 16;
        const float4 a0 = *(const float4*)(ap);
        const float4 a1 = *(const float4*)(ap + 4);
        const float4 a2 = *(const float4*)(ap + 8);
        const float4 a3 = *(const float4*)(ap + 12);
        v16bf af;
        af[0]  = (__bf16)a0.x; af[1]  = (__bf16)a0.y; af[2]  = (__bf16)a0.z; af[3]  = (__bf16)a0.w;
        af[4]  = (__bf16)a1.x; af[5]  = (__bf16)a1.y; af[6]  = (__bf16)a1.z; af[7]  = (__bf16)a1.w;
        af[8]  = (__bf16)a2.x; af[9]  = (__bf16)a2.y; af[10] = (__bf16)a2.z; af[11] = (__bf16)a2.w;
        af[12] = (__bf16)a3.x; af[13] = (__bf16)a3.y; af[14] = (__bf16)a3.z; af[15] = (__bf16)a3.w;

        // B fragments: one contiguous 32B load each from the packed buffer
        v16bf bfr[4];
        #pragma unroll
        for (int nn = 0; nn < 4; ++nn)
            bfr[nn] = *(const v16bf*)(Wp +
                (((size_t)kt * ntiles + (n0 >> 4) + nn) * 32 + lane) * 16);

        #pragma unroll
        for (int nn = 0; nn < 4; ++nn)
            acc[nn] = __builtin_amdgcn_wmma_f32_16x16x32_bf16(
                false, af, false, bfr[nn], (short)0, acc[nn], false, false);
    }

    #pragma unroll
    for (int nn = 0; nn < 4; ++nn) {
        #pragma unroll
        for (int r = 0; r < 8; ++r) {
            const int row = m0 + hi * 8 + r;      // D layout: VGPR r -> M = 8*hi + r
            const int col = n0 + nn * 16 + mr;    // column = lane&15 within 16-wide tile
            const float v = acc[nn][r] + bias[col];
            if (out_mode == 0) {
                C[(size_t)row * N + col] = v;
            } else {
                const int b  = row / NUM_KEYS_;
                const int kk = row - b * NUM_KEYS_;
                const int h  = col >> 5;
                const int d  = col & 31;
                C[(((size_t)b * HEADS_ + h) * NUM_KEYS_ + kk) * HEAD_DIM_ + d] = v;
            }
        }
    }
}

// ---------------------------------------------------------------------------
// Per-query: softmax over 16 points per head (16-wide shuffle groups in wave32),
// compute pixel-space sampling coords. One block = one (b,q), 128 threads =
// one thread per (h, l, p).
// ---------------------------------------------------------------------------
__global__ __launch_bounds__(128)
void prep_kernel(const float* __restrict__ off_raw, const float* __restrict__ attn_raw,
                 const float* __restrict__ refpts,  const int* __restrict__ shapes,
                 float* __restrict__ sx, float* __restrict__ sy, float* __restrict__ sw)
{
    const int bq = blockIdx.x;      // b*Q + q
    const int t  = threadIdx.x;     // 0..127 = h*16 + i
    const int h  = t >> 4;
    const int i  = t & 15;          // l*POINTS + p
    const int l  = i >> 2;

    // softmax over the 16 lanes of this head (contained in one 16-lane group)
    const float logit = attn_raw[(size_t)bq * 128 + t];
    float m = logit;
    #pragma unroll
    for (int s = 8; s >= 1; s >>= 1) m = fmaxf(m, __shfl_xor(m, s, 16));
    const float e = __expf(logit - m);
    float sum = e;
    #pragma unroll
    for (int s = 8; s >= 1; s >>= 1) sum += __shfl_xor(sum, s, 16);
    const float w = e / sum;

    const float ox = off_raw[(size_t)bq * 256 + t * 2 + 0];
    const float oy = off_raw[(size_t)bq * 256 + t * 2 + 1];
    const float rx = refpts[((size_t)bq * LEVELS_ + l) * 2 + 0];
    const float ry = refpts[((size_t)bq * LEVELS_ + l) * 2 + 1];
    const float Wl = (float)shapes[l * 2 + 1];
    const float Hl = (float)shapes[l * 2 + 0];
    const float x = (rx + ox / Wl) * Wl - 0.5f;   // pixel coords, matches reference
    const float y = (ry + oy / Hl) * Hl - 0.5f;

    const int b = bq / Q_;
    const int q = bq - b * Q_;
    const size_t o = (((size_t)b * HEADS_ + h) * Q_ + q) * 16 + i;
    sx[o] = x; sy[o] = y; sw[o] = w;
}

// ---------------------------------------------------------------------------
// Bilinear sampling + point-weighted sum. One wave per (b,h,q); lane = channel
// (HEAD_DIM = 32 = wave32). Each corner fetch is one contiguous 128B line;
// the whole projected value tensor (109 MB) lives in the 192 MB L2.
// ---------------------------------------------------------------------------
__global__ __launch_bounds__(256)
void sample_kernel(const float* __restrict__ vproj,
                   const float* __restrict__ sx, const float* __restrict__ sy,
                   const float* __restrict__ sw,
                   const int* __restrict__ shapes, const int* __restrict__ starts,
                   float* __restrict__ msda)
{
    const int lane = threadIdx.x & 31;
    const int wav  = threadIdx.x >> 5;
    const int bhq  = blockIdx.x * 8 + wav;        // (b*H + h)*Q + q

    int Hs[4], Ws[4], Ss[4];
    #pragma unroll
    for (int l = 0; l < 4; ++l) { Hs[l] = shapes[2*l]; Ws[l] = shapes[2*l+1]; Ss[l] = starts[l]; }

    const int    bh    = bhq / Q_;
    const size_t vbase = (size_t)bh * NUM_KEYS_ * HEAD_DIM_;
    const size_t p16   = (size_t)bhq * 16;

    float acc = 0.0f;
    for (int i = 0; i < 16; ++i) {
        const int l = i >> 2;
        const float x = sx[p16 + i];
        const float y = sy[p16 + i];
        const float w = sw[p16 + i];
        const float x0f = floorf(x), y0f = floorf(y);
        const float fx = x - x0f, fy = y - y0f;
        const int ix = (int)x0f, iy = (int)y0f;
        const int Wl = Ws[l], Hl = Hs[l], S = Ss[l];

        float v00 = 0.f, v01 = 0.f, v10 = 0.f, v11 = 0.f;
        const bool inx0 = (ix   >= 0) & (ix   < Wl);
        const bool inx1 = (ix+1 >= 0) & (ix+1 < Wl);
        const bool iny0 = (iy   >= 0) & (iy   < Hl);
        const bool iny1 = (iy+1 >= 0) & (iy+1 < Hl);
        if (inx0 & iny0) v00 = vproj[vbase + (size_t)(S + iy    *Wl + ix    ) * 32 + lane];
        if (inx1 & iny0) v01 = vproj[vbase + (size_t)(S + iy    *Wl + ix + 1) * 32 + lane];
        if (inx0 & iny1) v10 = vproj[vbase + (size_t)(S + (iy+1)*Wl + ix    ) * 32 + lane];
        if (inx1 & iny1) v11 = vproj[vbase + (size_t)(S + (iy+1)*Wl + ix + 1) * 32 + lane];

        acc += w * (v00 * (1.f-fx)*(1.f-fy) + v01 * fx*(1.f-fy)
                  + v10 * (1.f-fx)*fy       + v11 * fx*fy);
    }

    const int b   = bhq / (HEADS_ * Q_);
    const int rem = bhq - b * (HEADS_ * Q_);
    const int h   = rem / Q_;
    const int q   = rem - h * Q_;
    msda[((size_t)(b * Q_ + q)) * EMBED_ + h * HEAD_DIM_ + lane] = acc;
}

// ---------------------------------------------------------------------------
extern "C" void kernel_launch(void* const* d_in, const int* in_sizes, int n_in,
                              void* d_out, int out_size, void* d_ws, size_t ws_size,
                              hipStream_t stream)
{
    (void)in_sizes; (void)n_in; (void)out_size; (void)ws_size;

    const float* query  = (const float*)d_in[0];
    const float* refpts = (const float*)d_in[1];
    const float* value  = (const float*)d_in[2];
    const int*   shapes = (const int*)  d_in[3];
    const int*   starts = (const int*)  d_in[4];
    const float* W_off  = (const float*)d_in[5];
    const float* b_off  = (const float*)d_in[6];
    const float* W_attn = (const float*)d_in[7];
    const float* b_attn = (const float*)d_in[8];
    const float* W_val  = (const float*)d_in[9];
    const float* b_val  = (const float*)d_in[10];
    const float* W_out  = (const float*)d_in[11];
    const float* b_out  = (const float*)d_in[12];
    float* out = (float*)d_out;

    float* ws = (float*)d_ws;
    float* vproj    = ws; ws += (size_t)B_ * HEADS_ * NUM_KEYS_ * HEAD_DIM_; // 27,226,112
    float* off_raw  = ws; ws += (size_t)B_ * Q_ * 256;                       //  1,843,200
    float* attn_raw = ws; ws += (size_t)B_ * Q_ * 128;                       //    921,600
    float* sxp      = ws; ws += (size_t)B_ * Q_ * HEADS_ * 16;               //    921,600
    float* syp      = ws; ws += (size_t)B_ * Q_ * HEADS_ * 16;
    float* swp      = ws; ws += (size_t)B_ * Q_ * HEADS_ * 16;
    float* msda     = ws; ws += (size_t)B_ * Q_ * 256;                       //  1,843,200
    // packed bf16 weights (region is 64B-aligned: previous sizes sum to a 64B multiple)
    __bf16* Pval  = (__bf16*)ws;            // 256*256
    __bf16* Poff  = Pval  + 256 * 256;      // 256*256
    __bf16* Pattn = Poff  + 256 * 256;      // 256*128
    __bf16* Pout  = Pattn + 256 * 128;      // 256*256

    // 0) pack the four weight matrices into WMMA B-fragment order (bf16)
    pack_w_kernel<<<dim3(256 / 32, 256 / 16), 32, 0, stream>>>(W_val,  Pval,  256);
    pack_w_kernel<<<dim3(256 / 32, 256 / 16), 32, 0, stream>>>(W_off,  Poff,  256);
    pack_w_kernel<<<dim3(256 / 32, 128 / 16), 32, 0, stream>>>(W_attn, Pattn, 128);
    pack_w_kernel<<<dim3(256 / 32, 256 / 16), 32, 0, stream>>>(W_out,  Pout,  256);

    // 1) value projection -> (B, H, K, 32) scattered layout
    gemm_wmma_bf16<<<dim3(256 / 64, (B_ * NUM_KEYS_) / 16), 32, 0, stream>>>(
        value, Pval, b_val, vproj, 256, 256, /*mode=*/1);
    // 2) offset projection (B*Q, 256)
    gemm_wmma_bf16<<<dim3(256 / 64, (B_ * Q_) / 16), 32, 0, stream>>>(
        query, Poff, b_off, off_raw, 256, 256, /*mode=*/0);
    // 3) attention logits (B*Q, 128)
    gemm_wmma_bf16<<<dim3(128 / 64, (B_ * Q_) / 16), 32, 0, stream>>>(
        query, Pattn, b_attn, attn_raw, 128, 256, /*mode=*/0);
    // 4) softmax + sampling coordinates
    prep_kernel<<<B_ * Q_, 128, 0, stream>>>(off_raw, attn_raw, refpts, shapes,
                                             sxp, syp, swp);
    // 5) bilinear gather + weighted sum
    sample_kernel<<<(B_ * HEADS_ * Q_) / 8, 256, 0, stream>>>(vproj, sxp, syp, swp,
                                                              shapes, starts, msda);
    // 6) output projection -> d_out
    gemm_wmma_bf16<<<dim3(256 / 64, (B_ * Q_) / 16), 32, 0, stream>>>(
        msda, Pout, b_out, out, 256, 256, /*mode=*/0);
}
